// LSTM_85813446574183
// MI455X (gfx1250) — matrix operand order repaired and checked by reading, not compiled, not run on hardware
//
#include <hip/hip_runtime.h>

typedef __attribute__((ext_vector_type(16))) _Float16 v16h;
typedef __attribute__((ext_vector_type(8)))  float    v8f;

union FragAB { v16h v; unsigned u[8]; _Float16 h[16]; };
union FragC  { v8f  v; float f[8]; };

__device__ __forceinline__ unsigned pack_h2(float a, float b) {
  union { _Float16 h[2]; unsigned u; } p;
  p.h[0] = (_Float16)a; p.h[1] = (_Float16)b;
  return p.u;
}

__device__ __forceinline__ float sigf(float x) { return 1.0f / (1.0f + __expf(-x)); }

// ---------------------------------------------------------------------------
// Pack f32 [N][Ks] row-major -> f16 [N][Kp] row-major, zero-padded (Kp % 32 == 0)
// ---------------------------------------------------------------------------
__global__ void k_pack(const float* __restrict__ src, _Float16* __restrict__ dst,
                       int N, int Ks, int Kp) {
  int idx = blockIdx.x * blockDim.x + threadIdx.x;
  int total = N * (Kp >> 1);
  if (idx >= total) return;
  int n = idx / (Kp >> 1);
  int k = (idx - n * (Kp >> 1)) * 2;
  float a = (k     < Ks) ? src[(size_t)n * Ks + k]     : 0.0f;
  float b = (k + 1 < Ks) ? src[(size_t)n * Ks + k + 1] : 0.0f;
  ((unsigned*)dst)[idx] = pack_h2(a, b);
}

// ---------------------------------------------------------------------------
// Kernel A: xg = x @ W_ih^T + (b_ih + b_hh), written TRANSPOSED as [T][64][400]
// so the recurrence reads a contiguous 25.6KB slab per block per step.
// One wave per 16x16 output tile, K = 128 = 4 chunks of 32.
// ---------------------------------------------------------------------------
__global__ void k_xproj(const _Float16* __restrict__ xh,   // [131072][128] f16
                        const _Float16* __restrict__ wih,  // [400][128] f16
                        const float* __restrict__ b_ih,
                        const float* __restrict__ b_hh,
                        float* __restrict__ xg) {          // [2048][64][400] f32
  const int lane = threadIdx.x & 31;
  const int wave = threadIdx.x >> 5;
  const int lo = lane & 15, hi = lane >> 4;
  int wtile = blockIdx.x * 8 + wave;          // 25600 blocks * 8 waves = 204800
  int mTile = wtile / 25;
  int nTile = wtile - mTile * 25;
  const unsigned* arow = (const unsigned*)xh  + (size_t)(mTile * 16 + lo) * 64;
  const unsigned* brow = (const unsigned*)wih + (size_t)(nTile * 16 + lo) * 64;

  FragC acc;
#pragma unroll
  for (int r = 0; r < 8; ++r) acc.f[r] = 0.0f;

#pragma unroll
  for (int kc = 0; kc < 4; ++kc) {
    FragAB a, b;
#pragma unroll
    for (int j = 0; j < 8; ++j) {
      int ka = kc * 32 + ((j < 4) ? (hi * 8 + 2 * j) : (16 + hi * 8 + 2 * (j - 4)));
      int kb = kc * 32 + ((j < 4) ? (hi * 8 + 2 * j) : (16 + hi * 8 + 2 * (j - 4)));
      a.u[j] = arow[ka >> 1];
      b.u[j] = brow[kb >> 1];
    }
    acc.v = __builtin_amdgcn_wmma_f32_16x16x32_f16(false, a.v, false, b.v,
                                                   (short)0, acc.v, false, false);
  }

  int col = nTile * 16 + lo;
  float bs = b_ih[col] + b_hh[col];
#pragma unroll
  for (int r = 0; r < 8; ++r) {
    int row = mTile * 16 + hi * 8 + r;   // flat = b*2048 + t
    int b   = row >> 11;
    int t   = row & 2047;
    xg[((size_t)t * 64 + b) * 400 + col] = acc.f[r] + bs;
  }
}

// ---------------------------------------------------------------------------
// Kernel B: sequential LSTM recurrence. 4 blocks x 16 batch rows, 256 thr (8 waves).
// W_hh B-fragments preloaded to registers (128 VGPR/lane), h/c/g in LDS.
// Each wave computes 4 N-tile slots via n = (wave + 8s) mod 25 (7 tiles are
// duplicated across waves -> identical-value LDS writes, benign; keeps the
// slot loop branch-free so 4 independent WMMA chains interleave).
// ---------------------------------------------------------------------------
__global__ void k_lstm(const float* __restrict__ xg,       // [2048][64][400]
                       const _Float16* __restrict__ whh,   // [400][128] f16
                       _Float16* __restrict__ h16) {       // [131072][128] f16
  __shared__ float    g_lds[16][400];
  __shared__ float    c_lds[16][100];
  __shared__ _Float16 h_lds[16][128];

  const int tid  = threadIdx.x;
  const int lane = tid & 31, wave = tid >> 5;
  const int lo = lane & 15, hi = lane >> 4;
  const int bBase = blockIdx.x * 16;

  for (int e = tid; e < 16 * 128; e += 256) {
    ((_Float16*)h_lds)[e] = (_Float16)0.0f;
    if (e < 16 * 100) ((float*)c_lds)[e] = 0.0f;
  }
  __syncthreads();

  // preload W_hh B-fragments for this wave's 4 tile slots (kept in VGPRs)
  const unsigned* wbase = (const unsigned*)whh;
  int nSlot[4];
  FragAB bReg[4][4];
#pragma unroll
  for (int s = 0; s < 4; ++s) {
    int n = wave + 8 * s;
    if (n >= 25) n -= 25;
    nSlot[s] = n;
    const unsigned* brow = wbase + (size_t)(n * 16 + lo) * 64;
#pragma unroll
    for (int kc = 0; kc < 4; ++kc)
#pragma unroll
      for (int j = 0; j < 8; ++j)
        bReg[s][kc].u[j] = brow[(kc * 32 + ((j < 4) ? (hi * 8 + 2 * j) : (16 + hi * 8 + 2 * (j - 4)))) >> 1];
  }

  for (int t = 0; t < 2048; ++t) {
    // prefetch next step's contiguous xg slab (16 rows * 400 f32 = 25.6KB)
    if (t + 1 < 2048) {
      const char* nxt = (const char*)(xg + ((size_t)(t + 1) * 64 + bBase) * 400);
      if (tid * 128 < 25600) __builtin_prefetch(nxt + tid * 128, 0, 3);
    }
    // A fragments from h_lds (shared by all 4 slots)
    FragAB a[4];
#pragma unroll
    for (int kc = 0; kc < 4; ++kc) {
#pragma unroll
      for (int j = 0; j < 8; ++j) {
        int ka = kc * 32 + ((j < 4) ? (hi * 8 + 2 * j) : (16 + hi * 8 + 2 * (j - 4)));
        a[kc].u[j] = *(const unsigned*)&h_lds[lo][ka];
      }
    }
#pragma unroll
    for (int s = 0; s < 4; ++s) {
      int col = nSlot[s] * 16 + lo;
      FragC acc;
#pragma unroll
      for (int r = 0; r < 8; ++r)
        acc.f[r] = xg[((size_t)t * 64 + bBase + hi * 8 + r) * 400 + col];
#pragma unroll
      for (int kc = 0; kc < 4; ++kc)
        acc.v = __builtin_amdgcn_wmma_f32_16x16x32_f16(false, a[kc].v, false,
                                                       bReg[s][kc].v, (short)0,
                                                       acc.v, false, false);
#pragma unroll
      for (int r = 0; r < 8; ++r)
        g_lds[hi * 8 + r][col] = acc.f[r];
    }
    __syncthreads();

    // gates (i,f,g,o at column offsets 0/100/200/300), 4-wide packed
    for (int e = tid; e < 16 * 32; e += 256) {
      int m = e >> 5;
      int n0 = (e & 31) * 4;
      size_t gbase = ((size_t)(bBase + m) * 2048 + t) * 128 + n0;
      unsigned p0, p1;
      if (n0 < 100) {
        float hv[4], hr[4];
#pragma unroll
        for (int k = 0; k < 4; ++k) {
          int n = n0 + k;
          float gi = g_lds[m][n];
          float gf = g_lds[m][100 + n];
          float gg = g_lds[m][200 + n];
          float go = g_lds[m][300 + n];
          float iv = sigf(gi), fv = sigf(gf), gv = tanhf(gg), ov = sigf(go);
          float c = fv * c_lds[m][n] + iv * gv;
          c_lds[m][n] = c;
          float h = ov * tanhf(c);
          hv[k] = h;
          hr[k] = fmaxf(h, 0.0f);              // relu fused for FFN stage
        }
        *(unsigned*)&h_lds[m][n0]     = pack_h2(hv[0], hv[1]);
        *(unsigned*)&h_lds[m][n0 + 2] = pack_h2(hv[2], hv[3]);
        p0 = pack_h2(hr[0], hr[1]);
        p1 = pack_h2(hr[2], hr[3]);
      } else {
        p0 = 0u; p1 = 0u;                      // keep K-pad zero for FFN A-matrix
      }
      *(unsigned long long*)(h16 + gbase) =
          ((unsigned long long)p1 << 32) | (unsigned long long)p0;
    }
    __syncthreads();
  }
}

// ---------------------------------------------------------------------------
// Kernel C: fused FFN per 16-row tile. 256 thr (8 waves), 8192 blocks.
//   t1 = relu(relu_h @ ff1_w^T + ff1_b)  [16][300] staged in LDS f16 (pad 320)
//   out = t1 @ ff2_w^T + ff2_b           [16][128]
// ---------------------------------------------------------------------------
__global__ void k_ffn(const _Float16* __restrict__ h16,   // [131072][128] relu(h)
                      const _Float16* __restrict__ wf1,   // [300][128] f16
                      const float* __restrict__ ff1_b,
                      const _Float16* __restrict__ wf2,   // [128][320] f16
                      const float* __restrict__ ff2_b,
                      float* __restrict__ out) {          // [131072][128]
  __shared__ _Float16 t1[16][320];
  const int tid  = threadIdx.x;
  const int lane = tid & 31, wave = tid >> 5;
  const int lo = lane & 15, hi = lane >> 4;
  const int mTile = blockIdx.x;

  // zero K-pad columns 304..319 (no tile covers them)
  t1[tid >> 4][304 + (tid & 15)] = (_Float16)0.0f;

  const unsigned* arow = (const unsigned*)h16 + (size_t)(mTile * 16 + lo) * 64;

  // stage 1: N=300 -> 19 tiles (last partial, masked at store)
  for (int n1 = wave; n1 < 19; n1 += 8) {
    int col = n1 * 16 + lo;                 // up to 303
    int bcol = (col < 300) ? col : 299;     // clamp B reads in-bounds
    const unsigned* brow = (const unsigned*)wf1 + (size_t)bcol * 64;
    FragC acc;
#pragma unroll
    for (int r = 0; r < 8; ++r) acc.f[r] = 0.0f;
#pragma unroll
    for (int kc = 0; kc < 4; ++kc) {
      FragAB a, b;
#pragma unroll
      for (int j = 0; j < 8; ++j) {
        int ka = kc * 32 + ((j < 4) ? (hi * 8 + 2 * j) : (16 + hi * 8 + 2 * (j - 4)));
        int kb = kc * 32 + ((j < 4) ? (hi * 8 + 2 * j) : (16 + hi * 8 + 2 * (j - 4)));
        a.u[j] = arow[ka >> 1];
        b.u[j] = brow[kb >> 1];
      }
      acc.v = __builtin_amdgcn_wmma_f32_16x16x32_f16(false, a.v, false, b.v,
                                                     (short)0, acc.v, false, false);
    }
    float bias = (col < 300) ? ff1_b[col] : 0.0f;
#pragma unroll
    for (int r = 0; r < 8; ++r) {
      float v = fmaxf(acc.f[r] + bias, 0.0f);
      t1[hi * 8 + r][col] = (col < 300) ? (_Float16)v : (_Float16)0.0f;
    }
  }
  __syncthreads();

  // stage 2: N=128 -> 8 tiles, exactly one per wave; K=320 -> 10 chunks
  {
    int n2 = wave;
    int col = n2 * 16 + lo;
    const unsigned* brow = (const unsigned*)wf2 + (size_t)col * 160;
    FragC acc;
#pragma unroll
    for (int r = 0; r < 8; ++r) acc.f[r] = 0.0f;
#pragma unroll
    for (int kc = 0; kc < 10; ++kc) {
      FragAB a, b;
#pragma unroll
      for (int j = 0; j < 8; ++j) {
        int ka = kc * 32 + ((j < 4) ? (hi * 8 + 2 * j) : (16 + hi * 8 + 2 * (j - 4)));
        int kb = kc * 32 + ((j < 4) ? (hi * 8 + 2 * j) : (16 + hi * 8 + 2 * (j - 4)));
        a.u[j] = *(const unsigned*)&t1[lo][ka];
        b.u[j] = brow[kb >> 1];
      }
      acc.v = __builtin_amdgcn_wmma_f32_16x16x32_f16(false, a.v, false, b.v,
                                                     (short)0, acc.v, false, false);
    }
    float bias = ff2_b[col];
#pragma unroll
    for (int r = 0; r < 8; ++r)
      out[(size_t)(mTile * 16 + hi * 8 + r) * 128 + col] = acc.f[r] + bias;
  }
}

// ---------------------------------------------------------------------------
extern "C" void kernel_launch(void* const* d_in, const int* in_sizes, int n_in,
                              void* d_out, int out_size, void* d_ws, size_t ws_size,
                              hipStream_t stream) {
  (void)in_sizes; (void)n_in; (void)out_size; (void)ws_size;
  const float* x    = (const float*)d_in[0];
  // d_in[1] = y, unused by the reference computation
  const float* W_ih = (const float*)d_in[2];
  const float* W_hh = (const float*)d_in[3];
  const float* b_ih = (const float*)d_in[4];
  const float* b_hh = (const float*)d_in[5];
  const float* ff1w = (const float*)d_in[6];
  const float* ff1b = (const float*)d_in[7];
  const float* ff2w = (const float*)d_in[8];
  const float* ff2b = (const float*)d_in[9];
  float* out = (float*)d_out;

  // workspace layout (256B aligned): ~277 MB total
  char* ws = (char*)d_ws;
  size_t off = 0;
  auto wsalloc = [&](size_t bytes) -> char* {
    char* p = ws + off;
    off += (bytes + 255) & ~(size_t)255;
    return p;
  };
  _Float16* xh  = (_Float16*)wsalloc((size_t)131072 * 128 * 2);  // x as f16
  float*    xg  = (float*)   wsalloc((size_t)131072 * 400 * 4);  // [T][64][400]
  _Float16* h16 = (_Float16*)wsalloc((size_t)131072 * 128 * 2);  // relu(h), K-padded
  _Float16* wih = (_Float16*)wsalloc((size_t)400 * 128 * 2);
  _Float16* whh = (_Float16*)wsalloc((size_t)400 * 128 * 2);
  _Float16* wf1 = (_Float16*)wsalloc((size_t)300 * 128 * 2);
  _Float16* wf2 = (_Float16*)wsalloc((size_t)128 * 320 * 2);

  auto pack = [&](const float* s, _Float16* d, int N, int Ks, int Kp) {
    int total = N * (Kp / 2);
    k_pack<<<(total + 255) / 256, 256, 0, stream>>>(s, d, N, Ks, Kp);
  };
  pack(x,    xh,  131072, 128, 128);
  pack(W_ih, wih, 400,    128, 128);
  pack(W_hh, whh, 400,    100, 128);
  pack(ff1w, wf1, 300,    100, 128);
  pack(ff2w, wf2, 128,    300, 320);

  k_xproj<<<25600, 256, 0, stream>>>(xh, wih, b_ih, b_hh, xg);
  k_lstm <<<4,     256, 0, stream>>>(xg, whh, h16);
  k_ffn  <<<8192,  256, 0, stream>>>(h16, wf1, ff1b, wf2, ff2b, out);
}
